// TPBertaWithGates_84834194030958
// MI455X (gfx1250) — compile-verified, matrix operand order checked by reading, not run on hardware
//
#include <hip/hip_runtime.h>
#include <hip/hip_bf16.h>

// ---------------------------------------------------------------------------
// TPBertaWithGates forward for MI455X (gfx1250, wave32, WMMA bf16).
// GEMMs: v_wmma_f32_16x16x32_bf16 with ASYNC global->LDS double-buffered
// staging (global_load_async_to_lds_b128 / s_wait_asynccnt), LDS row stride
// padded to 160B to avoid 64-bank aliasing. Row-wise LN/GELU/sigmoid VALU
// kernels; per-column stats + 1-block finalize; per-row top-k approximated
// by deterministic bisection threshold.
// ---------------------------------------------------------------------------

typedef unsigned short u16;
typedef unsigned int   u32;
typedef __attribute__((ext_vector_type(4)))  unsigned int uint4v;
typedef __attribute__((ext_vector_type(16))) __bf16      v16bf;
typedef __attribute__((ext_vector_type(8)))  float       v8f;

static constexpr int kB = 4096;   // batch
static constexpr int kS = 1536;   // seq
static constexpr int kL = 2048;   // padded feature dim
static constexpr int kH = 1024;   // hidden
static constexpr int kMinTok = 460;           // max(1, int(0.3*1536))
static constexpr float kEps = 1e-7f;
static constexpr int kLdsStride = 80;         // 160B row stride: bank-conflict-free

__device__ __forceinline__ u16 f2bf(float f) {
  union { float f; u32 i; } c; c.f = f;
  u32 u = c.i + 0x7FFFu + ((c.i >> 16) & 1u);   // RNE
  return (u16)(u >> 16);
}

__device__ __forceinline__ float blkSum(float v, float* sred) {
  int t = threadIdx.x;
  sred[t] = v; __syncthreads();
  for (int s = 128; s > 0; s >>= 1) { if (t < s) sred[t] += sred[t + s]; __syncthreads(); }
  float r = sred[0]; __syncthreads();
  return r;
}

__device__ __forceinline__ int blkSumI(int v, int* sred) {
  int t = threadIdx.x;
  sred[t] = v; __syncthreads();
  for (int s = 128; s > 0; s >>= 1) { if (t < s) sred[t] += sred[t + s]; __syncthreads(); }
  int r = sred[0]; __syncthreads();
  return r;
}

// Direct global -> LDS async copy (16B per lane), ASYNCcnt-tracked.
__device__ __forceinline__ void async_ld_b128(u32 lds_addr, const void* gptr) {
  asm volatile("global_load_async_to_lds_b128 %0, %1, off"
               :: "v"(lds_addr), "v"(gptr) : "memory");
}
__device__ __forceinline__ void wait_async0() {
  asm volatile("s_wait_asynccnt 0x0" ::: "memory");
}

// ---------------------------------------------------------------------------
// f32 -> bf16 convert (weights)
// ---------------------------------------------------------------------------
__global__ __launch_bounds__(256) void cvt_bf16(const float* __restrict__ src,
                                                u16* __restrict__ dst, int n) {
  int i = blockIdx.x * 256 + threadIdx.x;
  if (i < n) dst[i] = f2bf(src[i]);
}

// ---------------------------------------------------------------------------
// Embed + LayerNorm over padded ids: x0 = LN(pad(ids)) * g + b  -> bf16 [B,L]
// ---------------------------------------------------------------------------
__global__ __launch_bounds__(256) void embed_ln(const int* __restrict__ ids,
                                                const float* __restrict__ g,
                                                const float* __restrict__ be,
                                                u16* __restrict__ out) {
  __shared__ float sred[256];
  const int b = blockIdx.x, t = threadIdx.x;
  const int* row = ids + (size_t)b * kS;
  float s = 0.f, s2 = 0.f;
  for (int i = t; i < kS; i += 256) { float v = (float)row[i]; s += v; s2 += v * v; }
  float sum   = blkSum(s,  sred);
  float sumsq = blkSum(s2, sred);
  float m   = sum / (float)kL;                       // pad columns contribute 0
  float var = sumsq / (float)kL - m * m;
  float inv = rsqrtf(var + 1e-5f);
  for (int i = t; i < kL; i += 256) {
    float v = (i < kS) ? (float)row[i] : 0.f;
    out[(size_t)b * kL + i] = f2bf((v - m) * inv * g[i] + be[i]);
  }
}

// ---------------------------------------------------------------------------
// GEMM: C[M,N] = A_bf16[M,K] @ W_bf16[N,K]^T + bias[N]
// 128x128 block tile, K-stage 64 (2 wmma sub-steps), double-buffered LDS
// filled by global_load_async_to_lds_b128. 8 waves x (2x4) wmma tiles.
// ---------------------------------------------------------------------------
union Frag { v16bf v; uint4v q[2]; };

__global__ __launch_bounds__(256) void gemm_bf16_wmma(
    const u16* __restrict__ A, const u16* __restrict__ W,
    const float* __restrict__ bias, float* __restrict__ Cf,
    u16* __restrict__ Cbf, int M, int N, int K) {
  __shared__ __align__(16) u16 sA[2][128][kLdsStride];
  __shared__ __align__(16) u16 sB[2][128][kLdsStride];

  const int tid  = threadIdx.x;
  const int lane = tid & 31;
  const int w    = tid >> 5;
  const int wm   = w & 3;        // 4 waves along M (32 rows each)
  const int wn   = w >> 2;       // 2 waves along N (64 cols each)
  const int m0   = (int)blockIdx.y * 128;
  const int n0   = (int)blockIdx.x * 128;

  v8f acc[2][4];
  const v8f vzero = {0.f, 0.f, 0.f, 0.f, 0.f, 0.f, 0.f, 0.f};
  for (int mi = 0; mi < 2; ++mi)
    for (int ni = 0; ni < 4; ++ni) acc[mi][ni] = vzero;

  const int khalf = (lane >> 4) * 8;  // per-lane K sub-chunk (ISA 16-bit layout)

  // Stage one 128x64 tile of A and W into LDS buffer `buf` via async DMA.
  // 128 rows * 8 chunks(16B) = 1024 chunks per matrix; 4 per thread each.
  auto stage = [&](int buf, int kk) {
    for (int it = 0; it < 4; ++it) {
      int idx = tid + it * 256;
      int r = idx >> 3, q = idx & 7;
      u32 la = (u32)(size_t)(void*)&sA[buf][r][q * 8];
      async_ld_b128(la, (const void*)(A + (size_t)(m0 + r) * K + kk + q * 8));
      u32 lb = (u32)(size_t)(void*)&sB[buf][r][q * 8];
      async_ld_b128(lb, (const void*)(W + (size_t)(n0 + r) * K + kk + q * 8));
    }
  };

  const int nstages = K >> 6;     // K/64
  stage(0, 0);
  for (int s = 0; s < nstages; ++s) {
    const int buf = s & 1;
    wait_async0();                 // this wave's pending fills are done
    __syncthreads();               // all waves' fills visible; prev buf dead
    if (s + 1 < nstages) stage(buf ^ 1, (s + 1) << 6);

    for (int ks = 0; ks < 2; ++ks) {
      const int kloc = ks * 32;
      Frag fa[2], fb[4];
      for (int mi = 0; mi < 2; ++mi) {
        int r = wm * 32 + mi * 16 + (lane & 15);
        fa[mi].q[0] = *(const uint4v*)&sA[buf][r][kloc + khalf];
        fa[mi].q[1] = *(const uint4v*)&sA[buf][r][kloc + 16 + khalf];
      }
      for (int ni = 0; ni < 4; ++ni) {
        int r = wn * 64 + ni * 16 + (lane & 15);
        fb[ni].q[0] = *(const uint4v*)&sB[buf][r][kloc + khalf];
        fb[ni].q[1] = *(const uint4v*)&sB[buf][r][kloc + 16 + khalf];
      }
      for (int mi = 0; mi < 2; ++mi)
        for (int ni = 0; ni < 4; ++ni)
          acc[mi][ni] = __builtin_amdgcn_wmma_f32_16x16x32_bf16(
              false, fa[mi].v, false, fb[ni].v, (short)0, acc[mi][ni], false, false);
    }
  }

  // Epilogue: D layout -> lane holds (m = r + 8*(lane>>4), n = lane&15) per tile.
  for (int mi = 0; mi < 2; ++mi) {
    int mbase = m0 + wm * 32 + mi * 16 + 8 * (lane >> 4);
    for (int ni = 0; ni < 4; ++ni) {
      int n = n0 + wn * 64 + ni * 16 + (lane & 15);
      float bn = bias[n];
      for (int r = 0; r < 8; ++r) {
        float v = acc[mi][ni][r] + bn;
        size_t off = (size_t)(mbase + r) * N + n;
        Cf[off] = v;
        if (Cbf) Cbf[off] = f2bf(v);
      }
    }
  }
}

// ---------------------------------------------------------------------------
// Row-wise LN + erf-GELU (+residual). resMode: 0 none, 1 += input, 2 += res[]
// ---------------------------------------------------------------------------
__global__ __launch_bounds__(256) void row_ln_act(
    const float* __restrict__ in, const float* __restrict__ gamma,
    const float* __restrict__ beta, const float* __restrict__ res, int resMode,
    float* __restrict__ outF, u16* __restrict__ outB, int N) {
  __shared__ float sred[256];
  const int b = blockIdx.x, t = threadIdx.x;
  const float* x = in + (size_t)b * N;
  float s = 0.f, s2 = 0.f;
  for (int i = t; i < N; i += 256) { float v = x[i]; s += v; s2 += v * v; }
  float sum   = blkSum(s,  sred);
  float sumsq = blkSum(s2, sred);
  float m   = sum / (float)N;
  float inv = rsqrtf(sumsq / (float)N - m * m + 1e-5f);
  for (int i = t; i < N; i += 256) {
    float v = x[i];
    float h = (v - m) * inv * gamma[i] + beta[i];
    float g = 0.5f * h * (1.f + erff(h * 0.70710678118f));
    if (resMode == 1) g += v;
    else if (resMode == 2) g += res[(size_t)b * N + i];
    if (outF) outF[(size_t)b * N + i] = g;
    outB[(size_t)b * N + i] = f2bf(g);
  }
}

// ---------------------------------------------------------------------------
// z = sigmoid(clip(mu, -5, 5))
// ---------------------------------------------------------------------------
__global__ __launch_bounds__(256) void z_kernel(const float* __restrict__ mu,
                                                float* __restrict__ z) {
  size_t i = (size_t)blockIdx.x * 256 + threadIdx.x;
  float v = mu[i];
  v = fminf(5.f, fmaxf(-5.f, v));
  z[i] = 1.f / (1.f + expf(-v));
}

// ---------------------------------------------------------------------------
// Per-column stats over batch: s1, s2, s11 (col i * col i+1), |diff| sum,
// binary-entropy sum. One thread per column; coalesced across threads.
// ---------------------------------------------------------------------------
__global__ __launch_bounds__(256) void colstats(
    const float* __restrict__ z, float* __restrict__ s1, float* __restrict__ s2,
    float* __restrict__ s11, float* __restrict__ dsum, float* __restrict__ ent) {
  int c = blockIdx.x * 256 + threadIdx.x;
  float a = 0.f, b2 = 0.f, c11 = 0.f, d = 0.f, e = 0.f;
  for (int b = 0; b < kB; ++b) {
    float v = z[(size_t)b * kL + c];
    a += v; b2 += v * v;
    e += -(v * logf(v + kEps) + (1.f - v) * logf(1.f - v + kEps));
    if (c < kL - 1) {
      float vn = z[(size_t)b * kL + c + 1];
      c11 += v * vn;
      d += fabsf(vn - v);
    }
  }
  s1[c] = a; s2[c] = b2; s11[c] = c11; dsum[c] = d; ent[c] = e;
}

// ---------------------------------------------------------------------------
// Finalize gating loss = R1 + R2 + consistency (single block).
// ---------------------------------------------------------------------------
__global__ __launch_bounds__(256) void finalize_loss(
    const float* __restrict__ s1, const float* __restrict__ s2,
    const float* __restrict__ s11, const float* __restrict__ dsum,
    const float* __restrict__ ent, const float* __restrict__ fe,
    float* __restrict__ loss_out) {
  __shared__ float sred[256];
  const int t = threadIdx.x;
  const float invB = 1.f / (float)kB;
  float cons = 0.f, r2 = 0.f, es = 0.f;
  for (int i = t; i < kL; i += 256) {
    cons += fabsf(s1[i] * invB - fe[i]);
    es += ent[i];
    if (i < kL - 1) {
      float mi = s1[i] * invB, mj = s1[i + 1] * invB;
      float cov = s11[i] * invB - mi * mj;
      float vi = fmaxf(s2[i] * invB - mi * mi, 0.f);
      float vj = fmaxf(s2[i + 1] * invB - mj * mj, 0.f);
      float corr = cov * rsqrtf(vi * vj + 1e-20f);
      if (corr > 0.5f) r2 += dsum[i];
    }
  }
  float consT = blkSum(cons, sred);
  float r2T   = blkSum(r2,   sred);
  float entT  = blkSum(es,   sred);
  if (t == 0) {
    float R1 = -0.001f * (entT / ((float)kB * (float)kL));
    float R2 = 0.001f * (r2T / ((float)kB * (float)(kL - 1)));
    float C  = 0.001f * (consT / (float)kL);
    loss_out[0] = R1 + R2 + C;
  }
}

// ---------------------------------------------------------------------------
// Per-row gating: mask = (z<0.5 & id!=0); if too many masked, protect ~top-k
// via deterministic bisection threshold on z in (0,1). Also emits z[:, :S].
// ---------------------------------------------------------------------------
__global__ __launch_bounds__(256) void gate_kernel(
    const float* __restrict__ z, const int* __restrict__ ids,
    float* __restrict__ gated_out, float* __restrict__ z_out) {
  __shared__ int sint[256];
  const int b = blockIdx.x, t = threadIdx.x;
  const float* zr = z + (size_t)b * kL;
  const int* idr = ids + (size_t)b * kS;

  int cnt = 0;
  for (int i = t; i < kS; i += 256)
    if (idr[i] != 0 && zr[i] < 0.5f) cnt++;
  int total = blkSumI(cnt, sint);
  bool need_fix = total > (kS - kMinTok);

  float thr = 2.f;  // nothing protected if !need_fix
  if (need_fix) {
    float lo = 0.f, hi = 1.f;
    for (int it = 0; it < 24; ++it) {
      float mid = 0.5f * (lo + hi);
      int c = 0;
      for (int i = t; i < kS; i += 256)
        if (zr[i] >= mid) c++;
      int cc = blkSumI(c, sint);
      if (cc > kMinTok) lo = mid; else hi = mid;
    }
    thr = lo;
  }

  for (int i = t; i < kS; i += 256) {
    int id = idr[i];
    float zv = zr[i];
    bool m = (zv < 0.5f) && (id != 0);
    if (need_fix && zv >= thr) m = false;   // protected (approx top-k)
    if (i == 0) m = false;                   // first token always kept
    gated_out[(size_t)b * kS + i] = m ? 0.f : (float)id;
    z_out[(size_t)b * kS + i] = zv;
  }
}

// ---------------------------------------------------------------------------
// Launch
// ---------------------------------------------------------------------------
extern "C" void kernel_launch(void* const* d_in, const int* in_sizes, int n_in,
                              void* d_out, int out_size, void* d_ws, size_t ws_size,
                              hipStream_t stream) {
  // Input order (setup_inputs): in_g, in_b, wv, bv, wo, bo, w2, b2, g1, be1,
  // w4, b4, g3, be3, w5, b5, g4, be4, w6, b6, g5, be5, w7, b7, g6, be6,
  // wout, bout, fe, input_ids
  const float* in_g = (const float*)d_in[0];
  const float* in_b = (const float*)d_in[1];
  const float* wv   = (const float*)d_in[2];
  const float* bv   = (const float*)d_in[3];
  const float* wo   = (const float*)d_in[4];
  const float* bo   = (const float*)d_in[5];
  const float* w2   = (const float*)d_in[6];
  const float* b2   = (const float*)d_in[7];
  const float* g1   = (const float*)d_in[8];
  const float* be1  = (const float*)d_in[9];
  const float* w4   = (const float*)d_in[10];
  const float* b4   = (const float*)d_in[11];
  const float* g3   = (const float*)d_in[12];
  const float* be3  = (const float*)d_in[13];
  const float* w5   = (const float*)d_in[14];
  const float* b5   = (const float*)d_in[15];
  const float* g4   = (const float*)d_in[16];
  const float* be4  = (const float*)d_in[17];
  const float* w6   = (const float*)d_in[18];
  const float* b6   = (const float*)d_in[19];
  const float* g5   = (const float*)d_in[20];
  const float* be5  = (const float*)d_in[21];
  const float* w7   = (const float*)d_in[22];
  const float* b7   = (const float*)d_in[23];
  const float* g6   = (const float*)d_in[24];
  const float* be6  = (const float*)d_in[25];
  const float* wout = (const float*)d_in[26];
  const float* bout = (const float*)d_in[27];
  const float* fe   = (const float*)d_in[28];
  const int*   ids  = (const int*)d_in[29];

  float* out = (float*)d_out;   // [gated B*S | loss 1 | z B*S]

  // ---- workspace carve (256B aligned) ----
  char* p = (char*)d_ws;
  auto carve = [&](size_t bytes) -> void* {
    void* r = (void*)p;
    p += (bytes + 255) & ~(size_t)255;
    return r;
  };
  u16* Wv   = (u16*)carve((size_t)kL * kL * 2);
  u16* Wo   = (u16*)carve((size_t)kL * kL * 2);
  u16* W2   = (u16*)carve((size_t)kH * kL * 2);
  u16* W4   = (u16*)carve((size_t)kH * kH * 2);
  u16* W5   = (u16*)carve((size_t)kH * kH * 2);
  u16* W6   = (u16*)carve((size_t)(kH / 2) * kH * 2);
  u16* W7   = (u16*)carve((size_t)(kH / 4) * (kH / 2) * 2);
  u16* Wout = (u16*)carve((size_t)kL * (kH / 4) * 2);
  u16* actA = (u16*)carve((size_t)kB * kL * 2);
  u16* actB = (u16*)carve((size_t)kB * kL * 2);
  float* pre = (float*)carve((size_t)kB * kL * 4);
  float* x2  = (float*)carve((size_t)kB * kH * 4);
  float* zb  = (float*)carve((size_t)kB * kL * 4);
  float* s1   = (float*)carve(kL * 4);
  float* s2   = (float*)carve(kL * 4);
  float* s11  = (float*)carve(kL * 4);
  float* dsum = (float*)carve(kL * 4);
  float* entb = (float*)carve(kL * 4);

  // ---- weight conversion f32 -> bf16 ----
  auto cvt = [&](const float* src, u16* dst, int n) {
    cvt_bf16<<<(n + 255) / 256, 256, 0, stream>>>(src, dst, n);
  };
  cvt(wv,   Wv,   kL * kL);
  cvt(wo,   Wo,   kL * kL);
  cvt(w2,   W2,   kH * kL);
  cvt(w4,   W4,   kH * kH);
  cvt(w5,   W5,   kH * kH);
  cvt(w6,   W6,   (kH / 2) * kH);
  cvt(w7,   W7,   (kH / 4) * (kH / 2));
  cvt(wout, Wout, kL * (kH / 4));

  auto gemm = [&](const u16* A, const u16* W, const float* bias, float* Cf,
                  u16* Cbf, int M, int N, int K) {
    dim3 grid(N / 128, M / 128);
    gemm_bf16_wmma<<<grid, 256, 0, stream>>>(A, W, bias, Cf, Cbf, M, N, K);
  };

  // ---- forward pipeline ----
  embed_ln<<<kB, 256, 0, stream>>>(ids, in_g, in_b, actA);                 // x0
  gemm(actA, Wv, bv, pre, actB, kB, kL, kL);                               // v_proj
  gemm(actB, Wo, bo, pre, actA, kB, kL, kL);                               // out_proj
  gemm(actA, W2, b2, pre, nullptr, kB, kH, kL);                            // id1
  row_ln_act<<<kB, 256, 0, stream>>>(pre, g1, be1, nullptr, 1, x2, actB, kH);   // x2 = gelu(LN)+id1
  gemm(actB, W4, b4, pre, nullptr, kB, kH, kH);
  row_ln_act<<<kB, 256, 0, stream>>>(pre, g3, be3, nullptr, 0, nullptr, actA, kH);
  gemm(actA, W5, b5, pre, nullptr, kB, kH, kH);
  row_ln_act<<<kB, 256, 0, stream>>>(pre, g4, be4, x2, 2, nullptr, actB, kH);   // + id2
  gemm(actB, W6, b6, pre, nullptr, kB, kH / 2, kH);
  row_ln_act<<<kB, 256, 0, stream>>>(pre, g5, be5, nullptr, 0, nullptr, actA, kH / 2);
  gemm(actA, W7, b7, pre, nullptr, kB, kH / 4, kH / 2);
  row_ln_act<<<kB, 256, 0, stream>>>(pre, g6, be6, nullptr, 0, nullptr, actB, kH / 4);
  gemm(actB, Wout, bout, pre, nullptr, kB, kL, kH / 4);                    // mu (pre-clip)

  z_kernel<<<(kB * kL) / 256, 256, 0, stream>>>(pre, zb);                  // z = sigmoid(clip)
  colstats<<<kL / 256, 256, 0, stream>>>(zb, s1, s2, s11, dsum, entb);
  finalize_loss<<<1, 256, 0, stream>>>(s1, s2, s11, dsum, entb, fe,
                                       out + (size_t)kB * kS);
  gate_kernel<<<kB, 256, 0, stream>>>(zb, ids, out,
                                      out + (size_t)kB * kS + 1);
}
